// MemoryK_52252572123192
// MI455X (gfx1250) — matrix operand order repaired and checked by reading, not compiled, not run on hardware
//
#include <hip/hip_runtime.h>

// ---------------- CDNA5 types ----------------
typedef __attribute__((ext_vector_type(16))) __bf16 v16bf;
typedef __attribute__((ext_vector_type(8)))  float  v8f;

union Frag {
    v16bf bf;
    unsigned int u[8];
    uint4 q[2];
};

static __device__ __forceinline__ unsigned int f2bf_rne(float x) {
    union { float f; unsigned int u; } c; c.f = x;
    unsigned int r = 0x7FFFu + ((c.u >> 16) & 1u);
    return (c.u + r) >> 16;
}
static __device__ __forceinline__ unsigned int pk_bf16(float lo, float hi) {
    return f2bf_rne(lo) | (f2bf_rne(hi) << 16);
}

#define NKEYS   65536
#define NCH     512
#define HW      49
#define MPAD    64
#define NSLAB   16
#define KEYS_PER_SLAB (NKEYS / NSLAB)      // 4096
#define NT_PER_SLAB   (KEYS_PER_SLAB / 16) // 256
#define TOPK    16
#define NCLS    100
#define ALPHA_F 0.1f
#define LDB     520   // LDS B-tile row stride in bf16 (512 + 8 pad -> conflict-free ds_load_b128)

// ---------------------------------------------------------------
// Kernel 0: one-time f32 -> bf16 conversion of m_keys.
// ---------------------------------------------------------------
__global__ __launch_bounds__(256) void kcvt_keys(const float* __restrict__ mk,
                                                 unsigned short* __restrict__ mkb) {
    size_t i = ((size_t)blockIdx.x * 256 + threadIdx.x) * 4;
    float4 x = *(const float4*)(mk + i);
    unsigned long long p = (unsigned long long)pk_bf16(x.x, x.y)
                         | ((unsigned long long)pk_bf16(x.z, x.w) << 32);
    *(unsigned long long*)(mkb + i) = p;
}

// ---------------------------------------------------------------
// Kernel 1: normalize queries (NCHW gather), wave per padded row.
// ---------------------------------------------------------------
__global__ __launch_bounds__(128) void knorm(const float* __restrict__ q,
                                             float* __restrict__ qn,
                                             unsigned short* __restrict__ qb) {
    int gwave = (blockIdx.x * blockDim.x + threadIdx.x) >> 5;
    int lane  = threadIdx.x & 31;
    int b = gwave >> 6;
    int m = gwave & 63;

    float v[16];
    float ss = 0.f;
    if (m < HW) {
        const float* src = q + (size_t)b * NCH * HW + m;   // + c*49
        #pragma unroll
        for (int j = 0; j < 16; ++j) {
            int c = lane + 32 * j;
            v[j] = src[(size_t)c * HW];
            ss += v[j] * v[j];
        }
    } else {
        #pragma unroll
        for (int j = 0; j < 16; ++j) v[j] = 0.f;
    }
    #pragma unroll
    for (int o = 16; o > 0; o >>= 1) ss += __shfl_xor(ss, o, 32);
    float scale = (m < HW) ? 1.0f / fmaxf(sqrtf(ss), 1e-12f) : 0.f;

    size_t rb = ((size_t)b * MPAD + m) * NCH;
    #pragma unroll
    for (int j = 0; j < 16; ++j) {
        int c = lane + 32 * j;
        float x = v[j] * scale;
        qn[rb + c] = x;
        qb[rb + c] = (unsigned short)f2bf_rne(x);
    }
}

// ---------------------------------------------------------------
// Kernel 2: per batch item, first memory index whose value == target.
// ---------------------------------------------------------------
__global__ void ksafety(const int* __restrict__ mv, const int* __restrict__ tgt,
                        int* __restrict__ safety) {
    __shared__ int smin;
    int b = blockIdx.x;
    if (threadIdx.x == 0) smin = 0x7FFFFFFF;
    __syncthreads();
    int t = tgt[b];
    for (int i = threadIdx.x; i < NKEYS; i += blockDim.x)
        if (mv[i] == t) atomicMin(&smin, i);
    __syncthreads();
    if (threadIdx.x == 0) safety[b] = smin;
}

// ---------------------------------------------------------------
// Kernel 3: WMMA GEMM + streaming top-16.
// - B tiles staged global->LDS with GLOBAL_LOAD_ASYNC_TO_LDS_B128
//   (ASYNCcnt, zero VGPR footprint), double-buffered in LDS.
// - B fragments double-buffered in registers; sched_group_barrier
//   pins the DS-read / WMMA interleave (4 DS up front, then 1 WMMA +
//   2 DS) so each WMMA overlaps the next fragment's LDS loads.
// ---------------------------------------------------------------
__global__ __launch_bounds__(128) void kgemm_topk(const unsigned short* __restrict__ qb,
                                                  const unsigned short* __restrict__ mkb,
                                                  float* __restrict__ topv,
                                                  int* __restrict__ topi) {
    const int slab = blockIdx.x;
    const int b    = blockIdx.y;
    const int tid  = threadIdx.x;
    const int w    = tid >> 5;
    const int lane = tid & 31;

    __shared__ alignas(16) unsigned short Bt[2][16 * LDB];
    __shared__ float sims[4][16][16];
    __shared__ float Ltv[MPAD][TOPK];
    __shared__ int   Lti[MPAD][TOPK];

    if (tid < MPAD) {
        #pragma unroll
        for (int j = 0; j < TOPK; ++j) { Ltv[tid][j] = -3.0e38f; Lti[tid][j] = 0; }
    }
    float tmin = -3.0e38f;
    int   tminpos = 0;

    // ---- register-resident A panel (16 rows x 512 K bf16) ----
    Frag a[16];
    {
        int row = b * MPAD + w * 16 + (lane & 15);
        const unsigned short* rp = qb + (size_t)row * NCH;
        int offA = (lane >= 16) ? 8 : 0;   // ISA A 16x32 half-wave K offset
        #pragma unroll
        for (int f = 0; f < 16; ++f) {
            a[f].q[0] = *(const uint4*)(rp + 32 * f + offA);
            a[f].q[1] = *(const uint4*)(rp + 32 * f + offA + 16);
        }
    }

    const int offB = (lane >= 16) ? 8 : 0;  // ISA B 32x16 half-wave K offset
    const int ncol = lane & 15;

    // per-thread staging slice: 8 x 16B units of the 16x512 bf16 tile.
    // unit i: row = (tid + 128i) >> 6 = (tid>>6) + 2i, kc = tid & 63.
    const int skc   = tid & 63;
    const int srow0 = tid >> 6;
    auto stage_async = [&](int nt, int buf) {
        int keybase = slab * KEYS_PER_SLAB + nt * 16;
        #pragma unroll
        for (int i = 0; i < 8; ++i) {
            int row = srow0 + 2 * i;
            unsigned long long gaddr =
                (unsigned long long)(mkb + (size_t)(keybase + row) * NCH + skc * 8);
            unsigned ldsoff =
                (unsigned)(unsigned long long)(&Bt[buf][row * LDB + skc * 8]);
            asm volatile("global_load_async_to_lds_b128 %0, %1, off"
                         :: "v"(ldsoff), "v"(gaddr) : "memory");
        }
    };

    stage_async(0, 0);
    asm volatile("s_wait_asynccnt 0x0" ::: "memory");
    __syncthreads();

    for (int nt = 0; nt < NT_PER_SLAB; ++nt) {
        const int cur = nt & 1;
        const int keybase = slab * KEYS_PER_SLAB + nt * 16;

        if (nt + 1 < NT_PER_SLAB) stage_async(nt + 1, cur ^ 1);  // DMA during compute

        const unsigned short* bp = &Bt[cur][ncol * LDB + offB];
        Frag bb[2];
        bb[0].q[0] = *(const uint4*)(bp);
        bb[0].q[1] = *(const uint4*)(bp + 16);
        v8f c = {};
        #pragma unroll
        for (int f = 0; f < 16; ++f) {
            if (f + 1 < 16) {   // issue next fragment's LDS loads before this WMMA
                bb[(f + 1) & 1].q[0] = *(const uint4*)(bp + 32 * (f + 1));
                bb[(f + 1) & 1].q[1] = *(const uint4*)(bp + 32 * (f + 1) + 16);
            }
            c = __builtin_amdgcn_wmma_f32_16x16x32_bf16(
                    false, a[f].bf, false, bb[f & 1].bf, (short)0, c, false, false);
        }
        // Pin the software pipeline: 4 DS reads (frags 0-1), then
        // (1 WMMA, 2 DS reads) x 15, final WMMA.  0x100 = DS read,
        // 0x008 = MFMA/WMMA.
        __builtin_amdgcn_sched_group_barrier(0x100, 4, 0);
        #pragma unroll
        for (int f = 0; f < 15; ++f) {
            __builtin_amdgcn_sched_group_barrier(0x008, 1, 0);
            __builtin_amdgcn_sched_group_barrier(0x100, 2, 0);
        }
        __builtin_amdgcn_sched_group_barrier(0x008, 1, 0);

        // C layout: VGPR r, lanes 0-15 -> row r, lanes 16-31 -> row r+8
        int rbase = (lane < 16) ? 0 : 8;
        #pragma unroll
        for (int r = 0; r < 8; ++r) sims[w][rbase + r][ncol] = c[r];
        __syncthreads();

        if (tid < HW) {  // owner thread per valid row: streaming top-16
            #pragma unroll
            for (int j = 0; j < 16; ++j) {
                float v = sims[tid >> 4][tid & 15][j];
                if (v > tmin) {
                    Ltv[tid][tminpos] = v;
                    Lti[tid][tminpos] = keybase + j;
                    tmin = Ltv[tid][0]; tminpos = 0;
                    #pragma unroll
                    for (int jj = 1; jj < TOPK; ++jj)
                        if (Ltv[tid][jj] < tmin) { tmin = Ltv[tid][jj]; tminpos = jj; }
                }
            }
        }

        asm volatile("s_wait_asynccnt 0x0" ::: "memory");  // next-tile DMA done
        __syncthreads();   // protects sims reuse + handoff of staged buffer
    }

    if (tid < HW) {
        size_t o = (((size_t)b * NSLAB + slab) * MPAD + tid) * TOPK;
        #pragma unroll
        for (int j = 0; j < TOPK; ++j) { topv[o + j] = Ltv[tid][j]; topi[o + j] = Lti[tid][j]; }
    }
}

// ---------------------------------------------------------------
// Kernel 4: merge slab lists, nearest-vote, safety column, margin loss.
// ---------------------------------------------------------------
__global__ __launch_bounds__(64) void kmerge_loss(const float* __restrict__ topv,
                                                  const int* __restrict__ topi,
                                                  const int* __restrict__ mv,
                                                  const int* __restrict__ tgt,
                                                  const int* __restrict__ safety,
                                                  const float* __restrict__ qn,
                                                  const float* __restrict__ mk,
                                                  float* __restrict__ perb,
                                                  float* __restrict__ dout) {
    int b = blockIdx.x;
    int t = threadIdx.x;

    __shared__ float Mtv[HW][TOPK];
    __shared__ int   Mti[HW][TOPK];
    __shared__ int   counts[NCLS + 1];
    __shared__ float rw[64], rn[64];

    for (int i = t; i < NCLS + 1; i += 64) counts[i] = 0;

    int nv = 0;
    if (t < HW) {
        float tmin = -3.0e38f; int tminpos = 0;
        #pragma unroll
        for (int j = 0; j < TOPK; ++j) { Mtv[t][j] = -3.0e38f; Mti[t][j] = 0; }
        for (int s = 0; s < NSLAB; ++s) {
            size_t o = (((size_t)b * NSLAB + s) * MPAD + t) * TOPK;
            for (int j = 0; j < TOPK; ++j) {
                float v = topv[o + j];
                if (v > tmin) {
                    Mtv[t][tminpos] = v; Mti[t][tminpos] = topi[o + j];
                    tmin = Mtv[t][0]; tminpos = 0;
                    #pragma unroll
                    for (int jj = 1; jj < TOPK; ++jj)
                        if (Mtv[t][jj] < tmin) { tmin = Mtv[t][jj]; tminpos = jj; }
                }
            }
        }
        float bv = -3.0e38f; int bi = 0;
        #pragma unroll
        for (int j = 0; j < TOPK; ++j)
            if (Mtv[t][j] > bv) { bv = Mtv[t][j]; bi = Mti[t][j]; }
        nv = mv[bi];
    }
    __syncthreads();
    if (t < HW) atomicAdd(&counts[nv + 1], 1);

    int tg = tgt[b];
    int sidx = safety[b];
    bool exists = (sidx < NKEYS);

    float lw = 0.f, ln = 0.f;
    if (t < HW) {
        float ssim = 0.f;
        if (exists) {
            const float* qr = qn + ((size_t)b * MPAD + t) * NCH;
            const float* kr = mk + (size_t)sidx * NCH;
            for (int k = 0; k < NCH; ++k) ssim += qr[k] * kr[k];
        }
        // masked-max with jnp's (sims * mask) zero semantics
        float pos = -3.0e38f, neg = -3.0e38f, negns = -3.0e38f;
        #pragma unroll
        for (int j = 0; j < TOPK; ++j) {
            float s = Mtv[t][j];
            bool m = (mv[Mti[t][j]] == tg);
            pos   = fmaxf(pos,   m ? s   : 0.f);
            neg   = fmaxf(neg,   m ? 0.f : s);
            negns = fmaxf(negns, m ? 0.f : s);
        }
        pos = fmaxf(pos, ssim);   // safety entry: mask true
        neg = fmaxf(neg, 0.f);    // safety entry contributes 0 to neg
        lw = fmaxf(neg - pos + ALPHA_F, 0.f);
        ln = fmaxf(negns + ALPHA_F, 0.f);
    }
    rw[t] = (t < HW) ? lw : 0.f;
    rn[t] = (t < HW) ? ln : 0.f;
    __syncthreads();

    if (t == 0) {
        float sw = 0.f, sn = 0.f;
        for (int i = 0; i < HW; ++i) { sw += rw[i]; sn += rn[i]; }
        float loss = exists ? (sw / (float)HW) : (sn / (float)HW);
        perb[b] = loss;
        int bestc = counts[0], besti = 0;
        for (int i = 1; i < NCLS + 1; ++i)
            if (counts[i] > bestc) { bestc = counts[i]; besti = i; }
        dout[b] = (float)(besti - 1);   // fetched value (numeric)
    }
}

__global__ void kfinal(const float* __restrict__ perb, float* __restrict__ dout) {
    if (threadIdx.x == 0 && blockIdx.x == 0) {
        float s = 0.f;
        for (int i = 0; i < 64; ++i) s += perb[i];
        dout[64] = s;
    }
}

// ---------------------------------------------------------------
extern "C" void kernel_launch(void* const* d_in, const int* in_sizes, int n_in,
                              void* d_out, int out_size, void* d_ws, size_t ws_size,
                              hipStream_t stream) {
    const float* queries = (const float*)d_in[0];   // (64,512,7,7) f32
    const int*   targets = (const int*)d_in[1];     // (64,) i32
    const float* m_keys  = (const float*)d_in[2];   // (65536,512) f32
    const int*   m_vals  = (const int*)d_in[3];     // (65536,) i32
    float* dout = (float*)d_out;                    // 64 fetched + 1 loss sum

    // scratch carving (all offsets 256B aligned)
    char* ws = (char*)d_ws;
    unsigned short* mkb    = (unsigned short*)ws;                          // 64 MB bf16 keys
    size_t off = (size_t)NKEYS * NCH * 2;
    float*          qn     = (float*)(ws + off);                           // 8 MB
    unsigned short* qbf    = (unsigned short*)(ws + off + 8388608);        // 4 MB
    float*          topv   = (float*)(ws + off + 8388608 + 4194304);       // 4 MB
    int*            topi   = (int*)  (ws + off + 8388608 + 2 * 4194304);   // 4 MB
    int*            safety = (int*)  (ws + off + 8388608 + 3 * 4194304);
    float*          perb   = (float*)(ws + off + 8388608 + 3 * 4194304 + 256);

    kcvt_keys<<<dim3(NKEYS * NCH / 4 / 256), dim3(256), 0, stream>>>(m_keys, mkb);
    knorm<<<dim3(1024), dim3(128), 0, stream>>>(queries, qn, qbf);
    ksafety<<<dim3(64), dim3(256), 0, stream>>>(m_vals, targets, safety);
    kgemm_topk<<<dim3(NSLAB, 64), dim3(128), 0, stream>>>(qbf, mkb, topv, topi);
    kmerge_loss<<<dim3(64), dim3(64), 0, stream>>>(topv, topi, m_vals, targets,
                                                   safety, qn, m_keys, perb, dout);
    kfinal<<<dim3(1), dim3(1), 0, stream>>>(perb, dout);
}
